// _SwinBlock_6588479832465
// MI455X (gfx1250) — compile-verified, matrix operand order checked by reading, not run on hardware
//
#include <hip/hip_runtime.h>
#include <hip/hip_bf16.h>

// ---------------------------------------------------------------------------
// Swin block for MI455X (gfx1250, wave32, WMMA f32_16x16x32_f16 everywhere).
// Pipeline: LN1+shift+window -> QKV WMMA GEMM -> per-(window,head) WMMA attn
//           -> proj WMMA GEMM (+window reverse +residual) -> LN2
//           -> MLP1 WMMA GEMM (+GELU) -> MLP2 WMMA GEMM (+residual).
// GEMMs: 128x128 block tile, 8 wave32s, double-buffered LDS, one barrier and
// 8 WMMAs per wave per K=32 chunk.
// ---------------------------------------------------------------------------

typedef __attribute__((ext_vector_type(16))) _Float16 v16h;
typedef __attribute__((ext_vector_type(8)))  _Float16 h8;
typedef __attribute__((ext_vector_type(8)))  float    v8f;

#define ROWS   131072        // B*H*W = 32*64*64
#define CDIM   384
#define HIDDIM 1536
#define NHEAD  12
#define HD     32

// ---- WMMA helpers ---------------------------------------------------------

__device__ __forceinline__ v8f wmma16(v16h a, v16h b, v8f c) {
  return __builtin_amdgcn_wmma_f32_16x16x32_f16(
      /*neg_a=*/false, a, /*neg_b=*/false, b,
      /*c_mod=*/(short)0, c, /*reuse_a=*/false, /*reuse_b=*/false);
}

// A/B fragment gather. `base` points at the start of this lane's source row
// (row-major, stride-1 along K) plus the K-chunk base. ISA 16-bit A layout:
//  lanes 0-15 / 16-31 both hold M(or N)=0..15; K interleave:
//  halves 0..7  -> k = 8*lane[4] + 0..7
//  halves 8..15 -> k = 8*lane[4] + 16..23
__device__ __forceinline__ v16h frag_row(const _Float16* base, int lane) {
  int o = ((lane >> 4) & 1) * 8;
  h8 lo = *(const h8*)(base + o);
  h8 hi = *(const h8*)(base + o + 16);
  v16h f;
#pragma unroll
  for (int j = 0; j < 8; ++j) { f[j] = lo[j]; f[8 + j] = hi[j]; }
  return f;
}

// Column gather (for V^T in attention): element h of the fragment for column
// `n` comes from src[k*ld + n] with the same K interleave as above.
__device__ __forceinline__ v16h frag_col(const _Float16* src, int ld, int n,
                                         int kBase, int lane) {
  int o = kBase + ((lane >> 4) & 1) * 8;
  v16h f;
#pragma unroll
  for (int j = 0; j < 8; ++j) {
    f[j]     = src[(o + j) * ld + n];
    f[8 + j] = src[(o + 16 + j) * ld + n];
  }
  return f;
}

// ---- LayerNorm (one wave32 per row, 12 elems/lane) ------------------------

__global__ __launch_bounds__(256)
void ln_kernel(const float* __restrict__ x, const float* __restrict__ w,
               const float* __restrict__ b, _Float16* __restrict__ out,
               int permute) {
  const int lane = threadIdx.x & 31;
  const int wv   = threadIdx.x >> 5;
  const int r    = blockIdx.x * 8 + wv;            // output row (window order if permute)
  const float* xr;
  if (permute) {
    // window-partitioned row -> shifted coords -> original coords (roll by -4)
    int bw = r >> 6, t = r & 63;
    int bb = bw >> 6, wwin = bw & 63;
    int hs = ((wwin >> 3) << 3) + (t >> 3);
    int ws = ((wwin & 7) << 3) + (t & 7);
    int ho = (hs + 4) & 63, wo = (ws + 4) & 63;
    xr = x + ((size_t)bb * 4096 + (size_t)ho * 64 + wo) * CDIM;
  } else {
    xr = x + (size_t)r * CDIM;
  }
  float vals[12], s = 0.f, ss = 0.f;
#pragma unroll
  for (int i = 0; i < 12; ++i) {
    float v = xr[lane + 32 * i];
    vals[i] = v; s += v; ss += v * v;
  }
#pragma unroll
  for (int off = 1; off < 32; off <<= 1) {
    s  += __shfl_xor(s,  off, 32);
    ss += __shfl_xor(ss, off, 32);
  }
  const float mean = s * (1.f / CDIM);
  const float var  = ss * (1.f / CDIM) - mean * mean;
  const float rstd = rsqrtf(var + 1e-5f);
#pragma unroll
  for (int i = 0; i < 12; ++i) {
    int c = lane + 32 * i;
    out[(size_t)r * CDIM + c] =
        (_Float16)((vals[i] - mean) * rstd * w[c] + b[c]);
  }
}

// ---- weight convert + transpose: w[K][N] (f32) -> wt[N][K] (f16) ----------

__global__ __launch_bounds__(256)
void cvt_t_kernel(const float* __restrict__ w, _Float16* __restrict__ wt,
                  int K, int N) {
  size_t i = (size_t)blockIdx.x * 256 + threadIdx.x;
  if (i < (size_t)K * N) {
    int kk = (int)(i / N), n = (int)(i % N);
    wt[(size_t)n * K + kk] = (_Float16)w[i];
  }
}

// ---- 128x128-tile WMMA GEMM: C = A[rows x K] * Bt[K x cols]^T -------------
// Block: 256 threads (8 wave32s). Wave w owns M-tiles {2(w&3), 2(w&3)+1} and
// N-tiles {4(w>>2)..+3}: 8 WMMAs / K-chunk, B fragments shared across the two
// A fragments. Double-buffered LDS -> one barrier per chunk; next chunk's
// global loads overlap current chunk's WMMAs.

template <typename Epi>
__global__ __launch_bounds__(256)
void gemm128_kernel(const _Float16* __restrict__ A, int lda,
                    const _Float16* __restrict__ Bt, int K, Epi epi) {
  __shared__ _Float16 As[2][128 * 32];
  __shared__ _Float16 Bs[2][128 * 32];
  const int t = threadIdx.x, lane = t & 31, wv = t >> 5;
  const int rowBase = blockIdx.x * 128;
  const int colBase = blockIdx.y * 128;
  const int mtb = (wv & 3) * 2;
  const int ntb = (wv >> 2) * 4;

  v8f acc[2][4] = {};

  // staging: 128 rows x 32 halves per matrix, 16 halves (2 x b128) per thread
  const int lr = t >> 1, lc = (t & 1) * 16;
  const _Float16* Ag = A  + (size_t)(rowBase + lr) * lda + lc;
  const _Float16* Bg = Bt + (size_t)(colBase + lr) * K   + lc;

  h8 a0 = *(const h8*)(Ag);     h8 a1 = *(const h8*)(Ag + 8);
  h8 b0 = *(const h8*)(Bg);     h8 b1 = *(const h8*)(Bg + 8);

  int buf = 0;
  for (int k0 = 0; k0 < K; k0 += 32) {
    *(h8*)(As[buf] + lr * 32 + lc)     = a0;
    *(h8*)(As[buf] + lr * 32 + lc + 8) = a1;
    *(h8*)(Bs[buf] + lr * 32 + lc)     = b0;
    *(h8*)(Bs[buf] + lr * 32 + lc + 8) = b1;
    __syncthreads();
    if (k0 + 32 < K) {                            // overlap next chunk's loads
      __builtin_prefetch(Ag + k0 + 64, 0, 1);     // gfx1250 global_prefetch_b8
      __builtin_prefetch(Bg + k0 + 64, 0, 1);
      a0 = *(const h8*)(Ag + k0 + 32);  a1 = *(const h8*)(Ag + k0 + 40);
      b0 = *(const h8*)(Bg + k0 + 32);  b1 = *(const h8*)(Bg + k0 + 40);
    }
    v16h af0 = frag_row(As[buf] + ((mtb + 0) * 16 + (lane & 15)) * 32, lane);
    v16h af1 = frag_row(As[buf] + ((mtb + 1) * 16 + (lane & 15)) * 32, lane);
#pragma unroll
    for (int j = 0; j < 4; ++j) {
      v16h bf = frag_row(Bs[buf] + ((ntb + j) * 16 + (lane & 15)) * 32, lane);
      acc[0][j] = wmma16(af0, bf, acc[0][j]);
      acc[1][j] = wmma16(af1, bf, acc[1][j]);
    }
    buf ^= 1;
  }
#pragma unroll
  for (int mi = 0; mi < 2; ++mi)
#pragma unroll
    for (int j = 0; j < 4; ++j)
#pragma unroll
      for (int i = 0; i < 8; ++i) {
        int row = rowBase + (mtb + mi) * 16 + i + 8 * (lane >> 4);
        int col = colBase + (ntb + j) * 16 + (lane & 15);
        epi(acc[mi][j][i], row, col);
      }
}

// ---- epilogues ------------------------------------------------------------

struct EpiQKV {   // split into q/k/v [win][head][n][d] f16; scale q
  const float* bias; _Float16* q; _Float16* k; _Float16* v; float scale;
  __device__ void operator()(float a, int row, int col) const {
    a += bias[col];
    int which = col / CDIM, rem = col - which * CDIM;
    int head = rem >> 5, d = rem & 31;
    int bw = row >> 6, n = row & 63;
    size_t dst = ((((size_t)bw * NHEAD + head) << 6) + n) * HD + d;
    if (which == 0) { q[dst] = (_Float16)(a * scale); }
    else if (which == 1) { k[dst] = (_Float16)a; }
    else { v[dst] = (_Float16)a; }
  }
};

struct EpiProj {  // +bias, window-reverse + un-roll, +residual(x) -> x2 (f32)
  const float* bias; const float* x; float* x2;
  __device__ void operator()(float a, int row, int col) const {
    a += bias[col];
    int bw = row >> 6, t = row & 63;
    int bb = bw >> 6, wwin = bw & 63;
    int hs = ((wwin >> 3) << 3) + (t >> 3);
    int ws = ((wwin & 7) << 3) + (t & 7);
    int ho = (hs + 4) & 63, wo = (ws + 4) & 63;
    size_t idx = ((size_t)bb * 4096 + (size_t)ho * 64 + wo) * CDIM + col;
    x2[idx] = a + x[idx];
  }
};

struct EpiMLP1 {  // +bias, exact GELU -> f16 hidden
  const float* b1; _Float16* mh;
  __device__ void operator()(float a, int row, int col) const {
    a += b1[col];
    a = 0.5f * a * (1.f + erff(a * 0.70710678118654752f));
    mh[(size_t)row * HIDDIM + col] = (_Float16)a;
  }
};

struct EpiMLP2 {  // +bias, +residual(x2) -> final output (f32)
  const float* b2; const float* x2; float* out;
  __device__ void operator()(float a, int row, int col) const {
    size_t idx = (size_t)row * CDIM + col;
    out[idx] = a + b2[col] + x2[idx];
  }
};

// ---- attention: one block per (window, head); 128 threads (4 waves) -------

__global__ __launch_bounds__(128)
void attn_kernel(const _Float16* __restrict__ q, const _Float16* __restrict__ k,
                 const _Float16* __restrict__ v, const float* __restrict__ rpb,
                 _Float16* __restrict__ o) {
  __shared__ _Float16 qs[64 * 32], ks[64 * 32], vs[64 * 32], ps[64 * 64];
  const int blk = blockIdx.x;
  const int head = blk % NHEAD, win = blk / NHEAD;
  const int wimg = win & 63;                     // window index within image
  const size_t base = ((size_t)win * NHEAD + head) * 64 * HD;
  const int t = threadIdx.x, lane = t & 31, wv = t >> 5;

  {  // stage q,k,v tiles (64x32 f16 each) into LDS
    int r = t >> 1, c = (t & 1) * 16;
    size_t g = base + (size_t)r * HD + c;
    *(h8*)(qs + r * 32 + c)     = *(const h8*)(q + g);
    *(h8*)(qs + r * 32 + c + 8) = *(const h8*)(q + g + 8);
    *(h8*)(ks + r * 32 + c)     = *(const h8*)(k + g);
    *(h8*)(ks + r * 32 + c + 8) = *(const h8*)(k + g + 8);
    *(h8*)(vs + r * 32 + c)     = *(const h8*)(v + g);
    *(h8*)(vs + r * 32 + c + 8) = *(const h8*)(v + g + 8);
  }
  __syncthreads();

  // S = q @ k^T for this wave's 16-row strip (K = hd = 32, one WMMA per tile)
  v8f s[4] = {};
  {
    v16h af = frag_row(qs + (wv * 16 + (lane & 15)) * 32, lane);
#pragma unroll
    for (int j = 0; j < 4; ++j) {
      v16h bf = frag_row(ks + (j * 16 + (lane & 15)) * 32, lane);
      s[j] = wmma16(af, bf, s[j]);
    }
  }

  // + relative position bias + shift mask (both computed analytically)
  const int whs = (wimg >> 3) << 3, wws = (wimg & 7) << 3;
#pragma unroll
  for (int j = 0; j < 4; ++j)
#pragma unroll
    for (int i = 0; i < 8; ++i) {
      int m = wv * 16 + i + 8 * (lane >> 4);
      int n = j * 16 + (lane & 15);
      int rm = m >> 3, cm = m & 7, rn = n >> 3, cn = n & 7;
      int idx = (rm - rn + 7) * 15 + (cm - cn + 7);
      float bias = rpb[idx * NHEAD + head];
      int hm = whs + rm, wm = wws + cm, hn = whs + rn, wn = wws + cn;
      int regm = ((hm < 56) ? 0 : (hm < 60) ? 1 : 2) * 3 +
                 ((wm < 56) ? 0 : (wm < 60) ? 1 : 2);
      int regn = ((hn < 56) ? 0 : (hn < 60) ? 1 : 2) * 3 +
                 ((wn < 56) ? 0 : (wn < 60) ? 1 : 2);
      s[j][i] += bias + ((regm != regn) ? -100.f : 0.f);
    }

  // row softmax: each row lives in one 16-lane half-wave (xor masks 1,2,4,8)
#pragma unroll
  for (int i = 0; i < 8; ++i) {
    float mx = s[0][i];
#pragma unroll
    for (int j = 1; j < 4; ++j) mx = fmaxf(mx, s[j][i]);
#pragma unroll
    for (int off = 1; off < 16; off <<= 1) mx = fmaxf(mx, __shfl_xor(mx, off, 32));
    float sum = 0.f;
#pragma unroll
    for (int j = 0; j < 4; ++j) { float e = __expf(s[j][i] - mx); s[j][i] = e; sum += e; }
#pragma unroll
    for (int off = 1; off < 16; off <<= 1) sum += __shfl_xor(sum, off, 32);
    float inv = 1.f / sum;
    int m = wv * 16 + i + 8 * (lane >> 4);
#pragma unroll
    for (int j = 0; j < 4; ++j)
      ps[m * 64 + j * 16 + (lane & 15)] = (_Float16)(s[j][i] * inv);
  }
  __syncthreads();

  // O = P @ v  (K = 64 in two chunks; 16x32 output per wave)
  v8f oacc[2] = {};
#pragma unroll
  for (int kc = 0; kc < 2; ++kc) {
    v16h pf = frag_row(ps + (wv * 16 + (lane & 15)) * 64 + kc * 32, lane);
#pragma unroll
    for (int nt = 0; nt < 2; ++nt) {
      v16h vf = frag_col(vs, 32, nt * 16 + (lane & 15), kc * 32, lane);
      oacc[nt] = wmma16(pf, vf, oacc[nt]);
    }
  }
#pragma unroll
  for (int nt = 0; nt < 2; ++nt)
#pragma unroll
    for (int i = 0; i < 8; ++i) {
      int m = wv * 16 + i + 8 * (lane >> 4);
      int d = nt * 16 + (lane & 15);
      o[(size_t)(win * 64 + m) * CDIM + head * HD + d] = (_Float16)oacc[nt][i];
    }
}

// ---- host-side launch -----------------------------------------------------

extern "C" void kernel_launch(void* const* d_in, const int* in_sizes, int n_in,
                              void* d_out, int out_size, void* d_ws, size_t ws_size,
                              hipStream_t stream) {
  const float* x      = (const float*)d_in[0];
  const float* ln1_w  = (const float*)d_in[1];
  const float* ln1_b  = (const float*)d_in[2];
  const float* ln2_w  = (const float*)d_in[3];
  const float* ln2_b  = (const float*)d_in[4];
  const float* qkv_w  = (const float*)d_in[5];
  const float* qkv_b  = (const float*)d_in[6];
  const float* proj_w = (const float*)d_in[7];
  const float* proj_b = (const float*)d_in[8];
  const float* rpb    = (const float*)d_in[9];
  const float* mlp_w1 = (const float*)d_in[10];
  const float* mlp_b1 = (const float*)d_in[11];
  const float* mlp_w2 = (const float*)d_in[12];
  const float* mlp_b2 = (const float*)d_in[13];
  float* out = (float*)d_out;

  // workspace layout (regions aliased across pipeline phases; peak ~710 MB)
  char* ws = (char*)d_ws;
  size_t off = 0;
  auto take = [&](size_t bytes) -> void* {
    void* p = ws + off;
    off += (bytes + 255) & ~(size_t)255;
    return p;
  };
  _Float16* qkvT  = (_Float16*)take((size_t)1152 * 384 * 2);
  _Float16* projT = (_Float16*)take((size_t)384 * 384 * 2);
  _Float16* w1T   = (_Float16*)take((size_t)1536 * 384 * 2);
  _Float16* w2T   = (_Float16*)take((size_t)384 * 1536 * 2);
  float*    x2    = (float*)   take((size_t)ROWS * CDIM * 4);
  _Float16* bufB  = (_Float16*)take((size_t)ROWS * CDIM * 2);       // q -> h2
  _Float16* D     = (_Float16*)take((size_t)ROWS * HIDDIM * 2);     // k|v|xw/o -> mh
  _Float16* kbuf  = D;
  _Float16* vbuf  = D + (size_t)ROWS * CDIM;
  _Float16* xw    = D + (size_t)2 * ROWS * CDIM;                    // LN1 out, later o
  _Float16* mh    = D;                                              // MLP hidden

  const float scale = 0.17677669529663687f;  // 32^-0.5

  // 1) f32 -> f16 transposed weights
  cvt_t_kernel<<<(384 * 1152 + 255) / 256, 256, 0, stream>>>(qkv_w,  qkvT, 384, 1152);
  cvt_t_kernel<<<(384 * 384  + 255) / 256, 256, 0, stream>>>(proj_w, projT, 384, 384);
  cvt_t_kernel<<<(384 * 1536 + 255) / 256, 256, 0, stream>>>(mlp_w1, w1T,  384, 1536);
  cvt_t_kernel<<<(1536 * 384 + 255) / 256, 256, 0, stream>>>(mlp_w2, w2T, 1536, 384);

  // 2) LN1 + shift + window partition -> xw (f16, window order)
  ln_kernel<<<ROWS / 8, 256, 0, stream>>>(x, ln1_w, ln1_b, xw, 1);

  // 3) QKV GEMM: [131072,384] x [384,1152]
  gemm128_kernel<EpiQKV><<<dim3(ROWS / 128, 9), 256, 0, stream>>>(
      xw, CDIM, qkvT, CDIM, EpiQKV{qkv_b, bufB, kbuf, vbuf, scale});

  // 4) windowed attention -> o (reuses xw region)
  attn_kernel<<<(ROWS / 64) * NHEAD, 128, 0, stream>>>(bufB, kbuf, vbuf, rpb, xw);

  // 5) proj GEMM + window reverse + residual -> x2 (f32)
  gemm128_kernel<EpiProj><<<dim3(ROWS / 128, 3), 256, 0, stream>>>(
      xw, CDIM, projT, CDIM, EpiProj{proj_b, x, x2});

  // 6) LN2 -> h2 (f16, reuses q region)
  ln_kernel<<<ROWS / 8, 256, 0, stream>>>(x2, ln2_w, ln2_b, bufB, 0);

  // 7) MLP1 GEMM + GELU -> mh (f16, reuses k/v/o region)
  gemm128_kernel<EpiMLP1><<<dim3(ROWS / 128, 12), 256, 0, stream>>>(
      bufB, CDIM, w1T, CDIM, EpiMLP1{mlp_b1, mh});

  // 8) MLP2 GEMM + residual -> out (f32)
  gemm128_kernel<EpiMLP2><<<dim3(ROWS / 128, 3), 256, 0, stream>>>(
      mh, HIDDIM, w2T, HIDDIM, EpiMLP2{mlp_b2, x2, out});
}
